// GPTLanguageModel_21414706937889
// MI455X (gfx1250) — matrix operand order repaired and checked by reading, not compiled, not run on hardware
//
#include <hip/hip_runtime.h>

// ---------------- CDNA5 WMMA types ----------------
typedef __attribute__((ext_vector_type(16))) __bf16 v16bf;   // 32 B: full fragment
typedef __attribute__((ext_vector_type(8)))  __bf16 bf16x8;  // 16 B: half fragment
typedef __attribute__((ext_vector_type(8)))  float  v8f;

// Model constants (from reference)
#define MB 2
#define MT 2048
#define MC 512
#define MH 8
#define ML 4
#define MV 32000
#define MMD 16
#define MNB 33
#define MD_HEAD 64
#define MF 2048          // 4*C
#define MM (MB*MT)       // 4096 rows

// ---------------- Embedding gather ----------------
__global__ __launch_bounds__(256) void embed_kernel(
    const int* __restrict__ tok, const float* __restrict__ emb,
    float* __restrict__ x, int C)
{
    int row = blockIdx.x;
    int t = tok[row];
    const float* e = emb + (size_t)t * C;
    float* xr = x + (size_t)row * C;
    for (int i = threadIdx.x; i < C; i += 256) xr[i] = e[i];
}

// ---------------- LayerNorm (one block per row) ----------------
__global__ __launch_bounds__(256) void layernorm_kernel(
    const float* __restrict__ x, const float* __restrict__ g,
    const float* __restrict__ b, float* __restrict__ out, int C)
{
    const int row = blockIdx.x;
    const int tid = threadIdx.x;
    const float* xr = x + (size_t)row * C;
    float* orow = out + (size_t)row * C;
    __shared__ float red[256];

    float s = 0.f;
    for (int i = tid; i < C; i += 256) s += xr[i];
    red[tid] = s; __syncthreads();
    for (int st = 128; st > 0; st >>= 1) {
        if (tid < st) red[tid] += red[tid + st];
        __syncthreads();
    }
    float mu = red[0] / (float)C;
    __syncthreads();

    float v = 0.f;
    for (int i = tid; i < C; i += 256) { float d = xr[i] - mu; v += d * d; }
    red[tid] = v; __syncthreads();
    for (int st = 128; st > 0; st >>= 1) {
        if (tid < st) red[tid] += red[tid + st];
        __syncthreads();
    }
    float rstd = rsqrtf(red[0] / (float)C + 1e-5f);

    for (int i = tid; i < C; i += 256)
        orow[i] = (xr[i] - mu) * rstd * g[i] + b[i];
}

// ---------------- WMMA GEMM: out = act(A @ W + bias) [+ residual] ----------------
// A: f32 [M x K] row-major (activations), W: f32 [K x N] row-major.
// LDS tiles are stored PRE-SWIZZLED into wave32 fragment order so each lane's
// 16 fragment bf16 are 32 contiguous bytes (2x ds_load_b128):
//   Aswz[buf][mt][lane][i] = A[mBase+mt*16+(lane&15)][kk+(i>=8?16:0)+(lane>=16?8:0)+(i&7)]
//   Wswz[buf][nt][lane][i] = W[kk+(i>=8?16:0)+(lane>=16?8:0)+(i&7)][nBase+nt*16+(lane&15)]
// Double-buffered + software-pipelined: next tile's global loads are issued
// before this tile's WMMAs; cvt+LDS-store happen after. One barrier / k-step.
// Macro tile 64(M) x 64(N), 8 waves (4x2), 2 WMMAs / wave / k-step.
#define GBM 64
#define GBN 64
#define GBK 32
__global__ __launch_bounds__(256) void gemm_bf16_wmma(
    const float* __restrict__ A, const float* __restrict__ W,
    const float* bias, const float* residual, float* out,
    int M, int N, int K, int relu)
{
    __shared__ __bf16 Aswz[2][4][32][16];     // 8 KB
    __shared__ __bf16 Wswz[2][4][32][16];     // 8 KB

    const int tid   = threadIdx.x;
    const int lane  = tid & 31;
    const int wave  = tid >> 5;            // 0..7
    const int hi    = lane >> 4;           // 0|1
    const int ln    = lane & 15;
    const int waveM = wave >> 1;           // 0..3  (16 rows each)
    const int waveN = wave & 1;            // 0..1  (32 cols each = 2 subtiles)
    const int mBase = blockIdx.y * GBM;
    const int nBase = blockIdx.x * GBN;

    // A-loader mapping: row = tid>>2 (0..63), c = (tid&3)*8 (0,8,16,24)
    const int a_row = tid >> 2;
    const int a_c   = (tid & 3) * 8;
    const int a_mt   = a_row >> 4;
    const int a_lane = (a_row & 15) + (((a_c & 15) >= 8) ? 16 : 0);
    const int a_i0   = (a_c >= 16) ? 8 : 0;
    // W-loader mapping: n = tid&63 (column), kgroup = tid>>6 (8 k's each)
    const int w_nloc = tid & 63;
    const int w_kg   = tid >> 6;
    const int w_nt   = w_nloc >> 4;
    const int w_lane = (w_nloc & 15) + ((w_kg & 1) ? 16 : 0);
    const int w_i0   = (w_kg >= 2) ? 8 : 0;

    // Running source pointers (advance by GBK / GBK*N per k-step)
    const float* ap = A + (size_t)(mBase + a_row) * K + a_c;
    const float* wp = W + (size_t)(w_kg * 8) * N + nBase + w_nloc;

    v8f acc0 = {}, acc1 = {};

    // ---- prologue: stage tile 0 into buffer 0 ----
    {
        const float4 f0 = *(const float4*)(ap);
        const float4 f1 = *(const float4*)(ap + 4);
        bf16x8 pa;
        pa[0] = (__bf16)f0.x; pa[1] = (__bf16)f0.y; pa[2] = (__bf16)f0.z; pa[3] = (__bf16)f0.w;
        pa[4] = (__bf16)f1.x; pa[5] = (__bf16)f1.y; pa[6] = (__bf16)f1.z; pa[7] = (__bf16)f1.w;
        *(bf16x8*)&Aswz[0][a_mt][a_lane][a_i0] = pa;
        bf16x8 pw;
        #pragma unroll
        for (int j = 0; j < 8; ++j) pw[j] = (__bf16)wp[(size_t)j * N];
        *(bf16x8*)&Wswz[0][w_nt][w_lane][w_i0] = pw;
    }

    const int nIter = K / GBK;
    int buf = 0;
    for (int it = 0; it < nIter; ++it) {
        __syncthreads();

        // 1) kick off next tile's global loads (long latency first)
        const bool haveNext = (it + 1) < nIter;
        float4 nf0 = {}, nf1 = {};
        float nw[8] = {};
        if (haveNext) {
            ap += GBK;
            wp += (size_t)GBK * N;
            nf0 = *(const float4*)(ap);
            nf1 = *(const float4*)(ap + 4);
            #pragma unroll
            for (int j = 0; j < 8; ++j) nw[j] = wp[(size_t)j * N];
            if ((it + 2) < nIter)
                __builtin_prefetch((const void*)(wp + (size_t)GBK * N), 0, 1);
        }

        // 2) fragments from current buffer (contiguous 32 B per lane) + WMMA
        v16bf a  = *(const v16bf*)(&Aswz[buf][waveM][lane][0]);
        v16bf b0 = *(const v16bf*)(&Wswz[buf][waveN * 2 + 0][lane][0]);
        v16bf b1 = *(const v16bf*)(&Wswz[buf][waveN * 2 + 1][lane][0]);
        acc0 = __builtin_amdgcn_wmma_f32_16x16x32_bf16(false, a, false, b0, (short)0, acc0, false, false);
        acc1 = __builtin_amdgcn_wmma_f32_16x16x32_bf16(false, a, false, b1, (short)0, acc1, false, false);

        // 3) convert + store next tile into the other buffer
        if (haveNext) {
            bf16x8 pa;
            pa[0] = (__bf16)nf0.x; pa[1] = (__bf16)nf0.y; pa[2] = (__bf16)nf0.z; pa[3] = (__bf16)nf0.w;
            pa[4] = (__bf16)nf1.x; pa[5] = (__bf16)nf1.y; pa[6] = (__bf16)nf1.z; pa[7] = (__bf16)nf1.w;
            *(bf16x8*)&Aswz[buf ^ 1][a_mt][a_lane][a_i0] = pa;
            bf16x8 pw;
            #pragma unroll
            for (int j = 0; j < 8; ++j) pw[j] = (__bf16)nw[j];
            *(bf16x8*)&Wswz[buf ^ 1][w_nt][w_lane][w_i0] = pw;
        }
        buf ^= 1;
    }

    // ---- epilogue: C/D layout m = r + 8*hi, n = lane&15 ----
    #pragma unroll
    for (int sub = 0; sub < 2; ++sub) {
        v8f acc = sub ? acc1 : acc0;
        #pragma unroll
        for (int r = 0; r < 8; ++r) {
            int m = mBase + waveM * 16 + r + (hi ? 8 : 0);
            int n = nBase + (waveN * 2 + sub) * 16 + ln;
            float v = acc[r];
            if (bias)     v += bias[n];
            if (relu)     v = fmaxf(v, 0.f);
            size_t idx = (size_t)m * N + n;
            if (residual) v += residual[idx];
            out[idx] = v;
        }
    }
}

// ---------------- Flash-style causal attention with rel-pos bias ----------------
// One wave per (b, h, 16-query tile). Q/K fragment loads are 4x global b128
// (their 16 bf16 come from 16 consecutive f32). Online softmax; P re-laid out
// C->A through 1 KB LDS; P@V with 4 WMMAs per 32-key chunk.
__global__ __launch_bounds__(32) void attn_kernel(
    const float* __restrict__ Q, const float* __restrict__ K,
    const float* __restrict__ V, const float* __restrict__ rpe, // (NB,H) for layer
    float* __restrict__ O, int T, int C, int H)
{
    const int lane = threadIdx.x;
    const int hi   = lane >> 4;
    const int ln   = lane & 15;
    const int qBase = blockIdx.x * 16;
    const int h     = blockIdx.y;
    const int b     = blockIdx.z;
    const float scale = 0.125f;            // 1/sqrt(64)

    const float* Qb = Q + (size_t)b * T * C + h * MD_HEAD;
    const float* Kb = K + (size_t)b * T * C + h * MD_HEAD;
    const float* Vb = V + (size_t)b * T * C + h * MD_HEAD;
    float*       Ob = O + (size_t)b * T * C + h * MD_HEAD;

    // Q A-fragments: lane row = qBase+ln; element i -> d = hi*16 + i
    // i 0..7 and 8..15 are two runs of 8 consecutive floats.
    v16bf aQ0, aQ1;
    {
        const float* qr = Qb + (size_t)(qBase + ln) * C + (hi ? 16 : 0);
        #pragma unroll
        for (int half = 0; half < 2; ++half) {
            const float4 f0 = *(const float4*)(qr + half * 8);
            const float4 f1 = *(const float4*)(qr + half * 8 + 4);
            const float4 g0 = *(const float4*)(qr + 32 + half * 8);
            const float4 g1 = *(const float4*)(qr + 32 + half * 8 + 4);
            aQ0[half*8+0]=(__bf16)f0.x; aQ0[half*8+1]=(__bf16)f0.y; aQ0[half*8+2]=(__bf16)f0.z; aQ0[half*8+3]=(__bf16)f0.w;
            aQ0[half*8+4]=(__bf16)f1.x; aQ0[half*8+5]=(__bf16)f1.y; aQ0[half*8+6]=(__bf16)f1.z; aQ0[half*8+7]=(__bf16)f1.w;
            aQ1[half*8+0]=(__bf16)g0.x; aQ1[half*8+1]=(__bf16)g0.y; aQ1[half*8+2]=(__bf16)g0.z; aQ1[half*8+3]=(__bf16)g0.w;
            aQ1[half*8+4]=(__bf16)g1.x; aQ1[half*8+5]=(__bf16)g1.y; aQ1[half*8+6]=(__bf16)g1.z; aQ1[half*8+7]=(__bf16)g1.w;
        }
    }

    v8f accO[4];
    #pragma unroll
    for (int dt = 0; dt < 4; ++dt) accO[dt] = (v8f){};
    float mstat[8], lstat[8];
    #pragma unroll
    for (int r = 0; r < 8; ++r) { mstat[r] = -__builtin_inff(); lstat[r] = 0.f; }

    __shared__ __bf16 Pl[16][32];          // P staging for C->A re-layout

    const int kend = qBase + 15;
    for (int kj = 0; kj <= kend; kj += 32) {
        // ---- scores S (16q x 32k) as two 16x16 subtiles ----
        v8f S[2];
        #pragma unroll
        for (int nt = 0; nt < 2; ++nt) {
            // B-frag: lane key = kj+nt*16+ln; element i -> d = hi*16 + i
            // => 16 consecutive floats at kr + hi*16 (and +32 for second half of D)
            const float* kr = Kb + (size_t)(kj + nt * 16 + ln) * C + (hi ? 16 : 0);
            v16bf bK0, bK1;
            #pragma unroll
            for (int qd = 0; qd < 4; ++qd) {
                const float4 f = *(const float4*)(kr + qd * 4);
                const float4 g = *(const float4*)(kr + 32 + qd * 4);
                bK0[qd*4+0]=(__bf16)f.x; bK0[qd*4+1]=(__bf16)f.y; bK0[qd*4+2]=(__bf16)f.z; bK0[qd*4+3]=(__bf16)f.w;
                bK1[qd*4+0]=(__bf16)g.x; bK1[qd*4+1]=(__bf16)g.y; bK1[qd*4+2]=(__bf16)g.z; bK1[qd*4+3]=(__bf16)g.w;
            }
            v8f c = {};
            c = __builtin_amdgcn_wmma_f32_16x16x32_bf16(false, aQ0, false, bK0, (short)0, c, false, false);
            c = __builtin_amdgcn_wmma_f32_16x16x32_bf16(false, aQ1, false, bK1, (short)0, c, false, false);
            S[nt] = c;
        }

        // scale + rpe bias + causal mask; per-row max
        float rowmax[8];
        #pragma unroll
        for (int r = 0; r < 8; ++r) rowmax[r] = -__builtin_inff();
        #pragma unroll
        for (int nt = 0; nt < 2; ++nt) {
            #pragma unroll
            for (int r = 0; r < 8; ++r) {
                int qrow = qBase + r + (hi ? 8 : 0);
                int kcol = kj + nt * 16 + ln;
                float s = S[nt][r] * scale;
                int rel = kcol - qrow + MMD;
                rel = rel < 0 ? 0 : (rel > 2 * MMD ? 2 * MMD : rel);
                s += rpe[rel * H + h];
                s = (kcol <= qrow) ? s : -__builtin_inff();
                S[nt][r] = s;
                rowmax[r] = fmaxf(rowmax[r], s);
            }
        }
        // reduce row max across the 16 lanes of each half-wave
        #pragma unroll
        for (int off = 1; off < 16; off <<= 1) {
            #pragma unroll
            for (int r = 0; r < 8; ++r)
                rowmax[r] = fmaxf(rowmax[r], __shfl_xor(rowmax[r], off, 32));
        }

        // online softmax update
        #pragma unroll
        for (int r = 0; r < 8; ++r) {
            float newmax = fmaxf(mstat[r], rowmax[r]);
            float corr = (mstat[r] == -__builtin_inff()) ? 0.f : __expf(mstat[r] - newmax);
            mstat[r] = newmax;
            #pragma unroll
            for (int dt = 0; dt < 4; ++dt) accO[dt][r] *= corr;
            float p0 = __expf(S[0][r] - newmax);
            float p1 = __expf(S[1][r] - newmax);
            S[0][r] = p0; S[1][r] = p1;
            float ps = p0 + p1;
            #pragma unroll
            for (int off = 1; off < 16; off <<= 1) ps += __shfl_xor(ps, off, 32);
            lstat[r] = lstat[r] * corr + ps;
        }

        // P (C-layout) -> LDS -> A-layout fragment
        #pragma unroll
        for (int nt = 0; nt < 2; ++nt)
            #pragma unroll
            for (int r = 0; r < 8; ++r)
                Pl[r + (hi ? 8 : 0)][nt * 16 + ln] = (__bf16)S[nt][r];
        __syncthreads();
        // A-frag of P: lane row = ln, contraction k = (i>=8?16:0)+hi*8+(i&7)
        // -> two contiguous 8-element runs in the LDS row
        v16bf aP;
        {
            const __bf16* pr = &Pl[ln][hi ? 8 : 0];
            bf16x8 lo  = *(const bf16x8*)(pr);
            bf16x8 hi8 = *(const bf16x8*)(pr + 16);
            #pragma unroll
            for (int j = 0; j < 8; ++j) { aP[j] = lo[j]; aP[8 + j] = hi8[j]; }
        }
        __syncthreads();

        // O += P @ V  (contraction over 32 keys, 4 D-subtiles)
        #pragma unroll
        for (int dt = 0; dt < 4; ++dt) {
            v16bf bV;
            #pragma unroll
            for (int i = 0; i < 16; ++i) {
                int key = kj + (i >= 8 ? 16 : 0) + (hi ? 8 : 0) + (i & 7);
                bV[i] = (__bf16)Vb[(size_t)key * C + dt * 16 + ln];
            }
            accO[dt] = __builtin_amdgcn_wmma_f32_16x16x32_bf16(
                           false, aP, false, bV, (short)0, accO[dt], false, false);
        }
    }

    // normalize and store
    #pragma unroll
    for (int dt = 0; dt < 4; ++dt)
        #pragma unroll
        for (int r = 0; r < 8; ++r) {
            int qrow = qBase + r + (hi ? 8 : 0);
            Ob[(size_t)qrow * C + dt * 16 + ln] = accO[dt][r] / lstat[r];
        }
}

// ---------------- Host launch ----------------
extern "C" void kernel_launch(void* const* d_in, const int* in_sizes, int n_in,
                              void* d_out, int out_size, void* d_ws, size_t ws_size,
                              hipStream_t stream) {
    (void)in_sizes; (void)n_in; (void)out_size; (void)ws_size;

    const int*   tokens  = (const int*)  d_in[0];
    const float* tok_emb = (const float*)d_in[1];
    const float* Wq      = (const float*)d_in[2];
    const float* Wk      = (const float*)d_in[3];
    const float* Wv      = (const float*)d_in[4];
    const float* Wo      = (const float*)d_in[5];
    const float* bo      = (const float*)d_in[6];
    const float* rpe     = (const float*)d_in[7];
    const float* ln1_g   = (const float*)d_in[8];
    const float* ln1_b   = (const float*)d_in[9];
    const float* ln2_g   = (const float*)d_in[10];
    const float* ln2_b   = (const float*)d_in[11];
    const float* W1      = (const float*)d_in[12];
    const float* b1      = (const float*)d_in[13];
    const float* W2      = (const float*)d_in[14];
    const float* b2      = (const float*)d_in[15];
    const float* lnf_g   = (const float*)d_in[16];
    const float* lnf_b   = (const float*)d_in[17];
    const float* Wout    = (const float*)d_in[18];
    const float* bout    = (const float*)d_in[19];

    const int M = MM;                       // 4096 rows
    const size_t RC = (size_t)M * MC;       // row-block of M*C floats

    float* x     = (float*)d_ws;            // [0,   RC)
    float* hbuf  = x + RC;                  // [RC, 2RC)   LN out; reused as attn out
    float* qb    = hbuf + RC;               // [2RC,3RC)
    float* kb    = qb + RC;                 // [3RC,4RC)
    float* vb    = kb + RC;                 // [4RC,5RC)
    /* extra  = vb + RC;                       [5RC,6RC) */
    float* m1    = qb;                      // spans q..extra = M * 4C floats

    embed_kernel<<<M, 256, 0, stream>>>(tokens, tok_emb, x, MC);

    const dim3 gC(MC / GBN, M / GBM);       // N=512 GEMMs
    const dim3 gF(MF / GBN, M / GBM);       // N=2048 GEMM
    const dim3 gV(MV / GBN, M / GBM);       // N=32000 GEMM
    const dim3 gAttn(MT / 16, MH, MB);

    for (int l = 0; l < ML; ++l) {
        layernorm_kernel<<<M, 256, 0, stream>>>(x, ln1_g + l * MC, ln1_b + l * MC, hbuf, MC);
        gemm_bf16_wmma<<<gC, 256, 0, stream>>>(hbuf, Wq + (size_t)l * MC * MC, nullptr, nullptr, qb, M, MC, MC, 0);
        gemm_bf16_wmma<<<gC, 256, 0, stream>>>(hbuf, Wk + (size_t)l * MC * MC, nullptr, nullptr, kb, M, MC, MC, 0);
        gemm_bf16_wmma<<<gC, 256, 0, stream>>>(hbuf, Wv + (size_t)l * MC * MC, nullptr, nullptr, vb, M, MC, MC, 0);
        attn_kernel<<<gAttn, 32, 0, stream>>>(qb, kb, vb, rpe + (size_t)l * MNB * MH, hbuf, MT, MC, MH);
        gemm_bf16_wmma<<<gC, 256, 0, stream>>>(hbuf, Wo + (size_t)l * MC * MC, bo + (size_t)l * MC, x, x, M, MC, MC, 0);
        layernorm_kernel<<<M, 256, 0, stream>>>(x, ln2_g + l * MC, ln2_b + l * MC, hbuf, MC);
        gemm_bf16_wmma<<<gF, 256, 0, stream>>>(hbuf, W1 + (size_t)l * MC * MF, b1 + (size_t)l * MF, nullptr, m1, M, MF, MC, 1);
        gemm_bf16_wmma<<<gC, 256, 0, stream>>>(m1, W2 + (size_t)l * MF * MC, b2 + (size_t)l * MC, x, x, M, MC, MF, 0);
    }

    layernorm_kernel<<<M, 256, 0, stream>>>(x, lnf_g, lnf_b, hbuf, MC);
    gemm_bf16_wmma<<<gV, 256, 0, stream>>>(hbuf, Wout, bout, nullptr, (float*)d_out, M, MV, MC, 0);
}
